// BitLinear_57492432224589
// MI455X (gfx1250) — matrix operand order repaired and checked by reading, not compiled, not run on hardware
//
#include <hip/hip_runtime.h>

// ---------------------------------------------------------------------------
// BitLinear: y[b,s,o] = scale[o] * sum_i x[b,s,i] * sign(w[o,i])
//   x: [4,2048,4096] f32  -> flattened M=8192, K=4096
//   w: [4096,4096]   f32  -> N=4096 rows (output features)
// bf16 WMMA GEMM (v_wmma_f32_16x16x32_bf16), fp32 accumulate, fp32 scale in
// epilogue. Double-buffered LDS filled with GLOBAL_LOAD_ASYNC_TO_LDS_B128
// (ASYNCcnt) overlapping the WMMAs. Block tile 128x256, wave tile 64x64
// (16 WMMAs per wave-stage; 0.5 KB LDS read per WMMA).
// Workspace layout (requires ~96.02 MB):
//   [0]      xbf   : 8192*4096 bf16 (64 MB)
//   [+64MB]  wsign : 4096*4096 bf16 (32 MB)
//   [+96MB]  scale : 4096 f32 (16 KB)
// ---------------------------------------------------------------------------

typedef __attribute__((ext_vector_type(16))) __bf16 v16bf;
typedef __attribute__((ext_vector_type(8)))  __bf16 v8bf;
typedef __attribute__((ext_vector_type(8)))  float  v8f;
typedef __attribute__((ext_vector_type(4)))  int    v4i;

#define D_IN   4096
#define D_OUT  4096
#define M_TOT  8192          // 4*2048
#define BK     32
#define BM     128           // block tile M
#define BN     256           // block tile N
#define LDS_STRIDE 40        // 32 + 8 bf16 pad (80B rows, 16B aligned)

#ifndef __has_builtin
#define __has_builtin(x) 0
#endif

#if __has_builtin(__builtin_amdgcn_global_load_async_to_lds_b128)
#define HAVE_ASYNC_LDS 1
#else
#define HAVE_ASYNC_LDS 0
#endif

#define GLOBAL_AS __attribute__((address_space(1)))
#define LDS_AS    __attribute__((address_space(3)))

// 16-byte global -> LDS copy. Async (memory->LDS direct, ASYNCcnt) when the
// builtin exists, otherwise synchronous through VGPRs.
// Builtin signature: (v4i AS1*, v4i AS3*, imm offset, imm cpol).
__device__ __forceinline__ void cp16_g2l(const unsigned short* g, unsigned short* l) {
#if HAVE_ASYNC_LDS
    __builtin_amdgcn_global_load_async_to_lds_b128(
        (GLOBAL_AS v4i*)(uintptr_t)g,
        (LDS_AS v4i*)(unsigned int)(uintptr_t)l,
        0 /*imm offset*/, 0 /*cpol*/);
#else
    *reinterpret_cast<uint4*>(l) = *reinterpret_cast<const uint4*>(g);
#endif
}

__device__ __forceinline__ void wait_async0() {
#if HAVE_ASYNC_LDS
#if __has_builtin(__builtin_amdgcn_s_wait_asynccnt)
    __builtin_amdgcn_s_wait_asynccnt(0);
#else
    asm volatile("s_wait_asynccnt 0x0" ::: "memory");
#endif
#endif
}

__device__ __forceinline__ void wait_ds0() {
    asm volatile("s_wait_dscnt 0x0" ::: "memory");
}

__device__ __forceinline__ unsigned short f32_to_bf16_rne(float f) {
    unsigned int u = __float_as_uint(f);
    unsigned int r = u + 0x7FFFu + ((u >> 16) & 1u);
    return (unsigned short)(r >> 16);
}

// -------------------------------------------------------------------
// Kernel 1: per-row mean(|w|) -> scale[o]; sign(w) -> bf16 {-1,0,+1}
// -------------------------------------------------------------------
__global__ void bitlinear_binarize_kernel(const float* __restrict__ w,
                                          unsigned short* __restrict__ wsign,
                                          float* __restrict__ scale) {
    const int o   = blockIdx.x;
    const int tid = threadIdx.x;
    const float* row = w + (size_t)o * D_IN;
    const int base = tid * 16;

    float acc = 0.0f;
    float vals[16];
#pragma unroll
    for (int j = 0; j < 4; ++j) {
        const float4 v = *reinterpret_cast<const float4*>(row + base + j * 4);
        vals[j * 4 + 0] = v.x; vals[j * 4 + 1] = v.y;
        vals[j * 4 + 2] = v.z; vals[j * 4 + 3] = v.w;
        acc += fabsf(v.x) + fabsf(v.y) + fabsf(v.z) + fabsf(v.w);
    }

#pragma unroll
    for (int off = 16; off > 0; off >>= 1)
        acc += __shfl_xor(acc, off, 32);

    __shared__ float red[8];
    const int wave = tid >> 5;
    if ((tid & 31) == 0) red[wave] = acc;
    __syncthreads();
    if (tid == 0) {
        float t = 0.0f;
#pragma unroll
        for (int i = 0; i < 8; ++i) t += red[i];
        scale[o] = t * (1.0f / (float)D_IN);
    }

    unsigned short s[16];
#pragma unroll
    for (int j = 0; j < 16; ++j) {
        const float f = vals[j];
        s[j] = (f > 0.0f) ? (unsigned short)0x3F80
             : (f < 0.0f) ? (unsigned short)0xBF80
                          : (unsigned short)0x0000;
    }
    unsigned short* dst = wsign + (size_t)o * D_IN + base;
#pragma unroll
    for (int j = 0; j < 4; ++j) {
        ushort4 p = make_ushort4(s[j*4+0], s[j*4+1], s[j*4+2], s[j*4+3]);
        *reinterpret_cast<ushort4*>(dst + j * 4) = p;
    }
}

// -------------------------------------------------------------------
// Kernel 2: x f32 -> bf16 (RNE), 8 elements per thread
// -------------------------------------------------------------------
__global__ void bitlinear_cvt_x_kernel(const float* __restrict__ x,
                                       unsigned short* __restrict__ xbf) {
    const size_t idx  = (size_t)blockIdx.x * blockDim.x + threadIdx.x;
    const size_t base = idx * 8;
    float4 a = *reinterpret_cast<const float4*>(x + base);
    float4 b = *reinterpret_cast<const float4*>(x + base + 4);
    ushort4 lo = make_ushort4(f32_to_bf16_rne(a.x), f32_to_bf16_rne(a.y),
                              f32_to_bf16_rne(a.z), f32_to_bf16_rne(a.w));
    ushort4 hi = make_ushort4(f32_to_bf16_rne(b.x), f32_to_bf16_rne(b.y),
                              f32_to_bf16_rne(b.z), f32_to_bf16_rne(b.w));
    *reinterpret_cast<ushort4*>(xbf + base)     = lo;
    *reinterpret_cast<ushort4*>(xbf + base + 4) = hi;
}

// -------------------------------------------------------------------
// Kernel 3: bf16 WMMA GEMM. grid = (N/256, M/128), block = 256 (8 waves).
// Wave (wm in 0..1, wn in 0..3) computes 64x64 = 4x4 WMMA tiles.
// Double-buffered LDS (60 KB), async global->LDS fill.
// -------------------------------------------------------------------
__global__ void __launch_bounds__(256)
bitlinear_gemm_kernel(const unsigned short* __restrict__ xbf,    // [8192,4096] bf16
                      const unsigned short* __restrict__ wsign,  // [4096,4096] bf16
                      const float* __restrict__ scale,           // [4096]
                      float* __restrict__ out) {                 // [8192,4096]
    __shared__ unsigned short As[2][BM * LDS_STRIDE];   // 2 x 10240 B
    __shared__ unsigned short Bs[2][BN * LDS_STRIDE];   // 2 x 20480 B

    const int tid    = threadIdx.x;
    const int waveId = tid >> 5;
    const int lane   = tid & 31;
    const int wm     = waveId >> 2;      // 0..1 : M sub-block of 64
    const int wn     = waveId & 3;       // 0..3 : N sub-block of 64
    const int r      = lane & 15;
    const int h      = lane >> 4;

    const int n0 = blockIdx.x * BN;
    const int m0 = blockIdx.y * BM;

    // fill chunk coordinates: A = 512 chunks (2/thread), B = 1024 (4/thread)
    int arow[2], acol[2], brow[4], bcol[4];
#pragma unroll
    for (int j = 0; j < 2; ++j) {
        const int c = tid + j * 256;
        arow[j] = c >> 2; acol[j] = (c & 3) * 8;
    }
#pragma unroll
    for (int j = 0; j < 4; ++j) {
        const int c = tid + j * 256;
        brow[j] = c >> 2; bcol[j] = (c & 3) * 8;
    }

    v8f acc[4][4];
#pragma unroll
    for (int mt = 0; mt < 4; ++mt)
#pragma unroll
        for (int nt = 0; nt < 4; ++nt)
            acc[mt][nt] = (v8f){0.f, 0.f, 0.f, 0.f, 0.f, 0.f, 0.f, 0.f};

    // prologue: stage 0 into buffer 0
#pragma unroll
    for (int j = 0; j < 2; ++j)
        cp16_g2l(xbf + (size_t)(m0 + arow[j]) * D_IN + acol[j],
                 &As[0][arow[j] * LDS_STRIDE + acol[j]]);
#pragma unroll
    for (int j = 0; j < 4; ++j)
        cp16_g2l(wsign + (size_t)(n0 + brow[j]) * D_IN + bcol[j],
                 &Bs[0][brow[j] * LDS_STRIDE + bcol[j]]);

    const int NIT = D_IN / BK;   // 128
    for (int it = 0; it < NIT; ++it) {
        const int cur = it & 1;

        // stage `it` fully in LDS (own loads), then block-wide visibility.
        wait_async0();
        __syncthreads();

        // kick off stage it+1 into the other buffer; safe: the barrier above
        // retired every wave's reads of that buffer (compute of it-1 ends
        // with s_wait_dscnt 0 before the barrier).
        if (it + 1 < NIT) {
            const int kn  = (it + 1) * BK;
            const int nxt = cur ^ 1;
#pragma unroll
            for (int j = 0; j < 2; ++j)
                cp16_g2l(xbf + (size_t)(m0 + arow[j]) * D_IN + kn + acol[j],
                         &As[nxt][arow[j] * LDS_STRIDE + acol[j]]);
#pragma unroll
            for (int j = 0; j < 4; ++j)
                cp16_g2l(wsign + (size_t)(n0 + brow[j]) * D_IN + kn + bcol[j],
                         &Bs[nxt][brow[j] * LDS_STRIDE + bcol[j]]);
        }

        // ---- compute on buffer `cur` ----
        // A fragments (16x32 bf16): lane holds row M = r,
        //   elems 0..7 = K[8h..8h+7], elems 8..15 = K[16+8h..16+8h+7]
        v16bf afrag[4];
#pragma unroll
        for (int mt = 0; mt < 4; ++mt) {
            const unsigned short* ap =
                &As[cur][(wm * 64 + mt * 16 + r) * LDS_STRIDE + h * 8];
            v8bf lo = *reinterpret_cast<const v8bf*>(ap);
            v8bf hi = *reinterpret_cast<const v8bf*>(ap + 16);
            afrag[mt] = __builtin_shufflevector(lo, hi, 0,1,2,3,4,5,6,7,
                                                        8,9,10,11,12,13,14,15);
        }
        // B fragments (32x16 bf16): lane holds column N = r,
        //   elems 0..7 = K[8h..8h+7], elems 8..15 = K[16+8h..16+8h+7]
        v16bf bfrag[4];
#pragma unroll
        for (int nt = 0; nt < 4; ++nt) {
            const unsigned short* bp =
                &Bs[cur][(wn * 64 + nt * 16 + r) * LDS_STRIDE + h * 8];
            v8bf lo = *reinterpret_cast<const v8bf*>(bp);
            v8bf hi = *reinterpret_cast<const v8bf*>(bp + 16);
            bfrag[nt] = __builtin_shufflevector(lo, hi, 0,1,2,3,4,5,6,7,
                                                        8,9,10,11,12,13,14,15);
        }

#pragma unroll
        for (int mt = 0; mt < 4; ++mt)
#pragma unroll
            for (int nt = 0; nt < 4; ++nt)
                acc[mt][nt] = __builtin_amdgcn_wmma_f32_16x16x32_bf16(
                    false, afrag[mt], false, bfrag[nt],
                    (short)0, acc[mt][nt], false, false);

        // all LDS reads of buffer `cur` retired before the next barrier
        wait_ds0();
    }

    // epilogue: multiply by per-output-feature fp32 scale, store.
    // C/D layout: VGPR v holds M = 16*mt + v + 8*h, lane r -> N = 16*nt + r.
#pragma unroll
    for (int nt = 0; nt < 4; ++nt) {
        const int n  = n0 + wn * 64 + nt * 16 + r;
        const float sc = scale[n];
#pragma unroll
        for (int mt = 0; mt < 4; ++mt) {
            const int mbase = m0 + wm * 64 + mt * 16 + h * 8;
#pragma unroll
            for (int v = 0; v < 8; ++v)
                out[(size_t)(mbase + v) * D_OUT + n] = acc[mt][nt][v] * sc;
        }
    }
}

// -------------------------------------------------------------------
extern "C" void kernel_launch(void* const* d_in, const int* in_sizes, int n_in,
                              void* d_out, int out_size, void* d_ws, size_t ws_size,
                              hipStream_t stream) {
    const float* x = (const float*)d_in[0];   // [4,2048,4096]
    const float* w = (const float*)d_in[1];   // [4096,4096]
    float* out = (float*)d_out;

    unsigned short* xbf   = (unsigned short*)d_ws;                       // 64 MB
    unsigned short* wsign = xbf + (size_t)M_TOT * D_IN;                  // 32 MB
    float*          scale = (float*)(wsign + (size_t)D_OUT * D_IN);      // 16 KB

    bitlinear_binarize_kernel<<<dim3(D_OUT), dim3(256), 0, stream>>>(w, wsign, scale);

    const int cvt_blocks = (int)(((size_t)M_TOT * D_IN) / (8 * 256));
    bitlinear_cvt_x_kernel<<<dim3(cvt_blocks), dim3(256), 0, stream>>>(x, xbf);

    bitlinear_gemm_kernel<<<dim3(D_OUT / BN, M_TOT / BM), dim3(256), 0, stream>>>(
        xbf, wsign, scale, out);
    (void)in_sizes; (void)n_in; (void)out_size; (void)ws_size;
}
